// GMT_31413390803186
// MI455X (gfx1250) — compile-verified, matrix operand order checked by reading, not run on hardware
//
#include <hip/hip_runtime.h>
#include <hip/hip_bf16.h>

// ---------- types ----------
typedef __attribute__((ext_vector_type(16))) __bf16          v16bf;
typedef __attribute__((ext_vector_type(16))) unsigned short  v16u;
typedef __attribute__((ext_vector_type(8)))  unsigned short  u16x8;
typedef __attribute__((ext_vector_type(8)))  float           v8f;

#define B_  64
#define S_  512
#define D_  40
#define H_  512
#define BS_ (B_ * S_)
#define NBLK_ 16   // persistent blocks per recurrence (8 waves each -> 128 tile-waves)

// ---------- small helpers ----------
static __device__ __forceinline__ unsigned short f2bf(float f) {
    unsigned u = __float_as_uint(f);
    unsigned r = u + 0x7FFFu + ((u >> 16) & 1u);   // RNE
    return (unsigned short)(r >> 16);
}
static __device__ __forceinline__ float bf2f(unsigned short h) {
    return __uint_as_float(((unsigned)h) << 16);
}
static __device__ __forceinline__ float sigmoidf(float x) {
    return 1.0f / (1.0f + expf(-x));
}

// Load one 16x32 bf16 WMMA operand tile from a row-major bf16 matrix.
// Used for both A (lane = M row) and B (lane = N column of W, W stored [N,K]).
// Per ISA 7.12.2: lanes 0-15 hold K {k0..k0+7, k0+16..k0+23},
//                 lanes 16-31 hold K {k0+8..k0+15, k0+24..k0+31}.
static __device__ __forceinline__ v16bf load_tile_bf16(
    const unsigned short* __restrict__ base, int ld, int row0, int k0, int lane)
{
    int r  = row0 + (lane & 15);
    int kb = k0 + ((lane >> 4) << 3);
    const unsigned short* p = base + (size_t)r * ld + kb;
    u16x8 lo = *(const u16x8*)(p);
    u16x8 hi = *(const u16x8*)(p + 16);
    v16u t;
#pragma unroll
    for (int i = 0; i < 8; ++i) { t[i] = lo[i]; t[i + 8] = hi[i]; }
    return __builtin_bit_cast(v16bf, t);
}

static __device__ __forceinline__ v8f wmma_bf16(v16bf a, v16bf b, v8f c) {
    return __builtin_amdgcn_wmma_f32_16x16x32_bf16(
        false, a, false, b, (short)0, c, false, false);
}

// Inter-block step barrier: device-scope release, monotonically increasing
// counter (never reset within a call), spin with s_sleep, acquire on exit.
static __device__ __forceinline__ void step_barrier(unsigned* bar, int t) {
    __threadfence();            // release: h/c/o writes visible device-wide
    __syncthreads();            // all waves in this block arrived
    if (threadIdx.x == 0) {
        atomicAdd(bar, 1u);
        const unsigned tgt = (unsigned)NBLK_ * (unsigned)(t + 1);
        while (atomicAdd(bar, 0u) < tgt) __builtin_amdgcn_s_sleep(1);
    }
    __syncthreads();
    __threadfence();            // acquire: observe other blocks' writes
}

// ---------- prep kernels ----------
__global__ void cvt_pad_kernel(const float* __restrict__ src,
                               unsigned short* __restrict__ dst,
                               int N, int K, int Kpad) {
    int idx = blockIdx.x * blockDim.x + threadIdx.x;
    if (idx >= N * Kpad) return;
    int c = idx % Kpad, r = idx / Kpad;
    float v = (c < K) ? src[(size_t)r * K + c] : 0.0f;
    dst[idx] = f2bf(v);
}

// x [B,S,D] f32 -> xbG/xbL [S,B,64] bf16 (cols >= D zero; xbL col 40 filled later)
__global__ void build_x_kernel(const float* __restrict__ x,
                               unsigned short* __restrict__ xbG,
                               unsigned short* __restrict__ xbL) {
    int idx = blockIdx.x * blockDim.x + threadIdx.x;
    if (idx >= BS_ * 64) return;
    int c = idx & 63, row = idx >> 6;
    int b = row & (B_ - 1), s = row >> 6;
    float v = (c < D_) ? x[((size_t)b * S_ + s) * D_ + c] : 0.0f;
    unsigned short h = f2bf(v);
    xbG[idx] = h;
    xbL[idx] = h;
}

__global__ void zero32_kernel(unsigned int* __restrict__ p, int n) {
    int i = blockIdx.x * blockDim.x + threadIdx.x;
    if (i < n) p[i] = 0u;
}

// ---------- persistent GRU: 16 blocks x 8 waves, full 512-step scan ----------
__global__ __launch_bounds__(256) void gru_persist_kernel(
    const unsigned short* __restrict__ xb,     // [S,64,64] bf16 padded inputs
    unsigned short* __restrict__ h0,           // [64,512] bf16 (parity 0)
    unsigned short* __restrict__ h1,           // [64,512] bf16 (parity 1)
    float* __restrict__ o1,                    // [S,64,512] f32
    const unsigned short* __restrict__ Wih,    // [1536,64] bf16
    const unsigned short* __restrict__ Whh,    // [1536,512] bf16
    const float* __restrict__ bih,             // [1536]
    const float* __restrict__ bhh,             // [1536]
    unsigned* __restrict__ bar)
{
    const int lane = threadIdx.x & 31;
    const int wv   = threadIdx.x >> 5;
    const int idx  = blockIdx.x * 8 + wv;      // 0..127 tile id
    const int m0   = (idx >> 5) << 4;          // batch tile
    const int n0   = (idx & 31) << 4;          // hidden tile

    const int col   = n0 + (lane & 15);
    const float br_ = bih[col]          + bhh[col];
    const float bz_ = bih[H_ + col]     + bhh[H_ + col];
    const float bnx = bih[2 * H_ + col];
    const float bnh = bhh[2 * H_ + col];
    const int mbase = m0 + ((lane >> 4) << 3);

    for (int t = 0; t < S_; ++t) {
        const unsigned short* hprev = (t & 1) ? h1 : h0;
        unsigned short*       hnext = (t & 1) ? h0 : h1;
        const unsigned short* xt = xb + (size_t)t * (B_ * 64);
        float* o1t = o1 + (size_t)t * (B_ * H_);

        // prefetch next step's input tile (global_prefetch_b8)
        if (t + 1 < S_)
            __builtin_prefetch(xb + (size_t)(t + 1) * (B_ * 64) + lane * 128, 0, 0);

        v8f cr  = {0,0,0,0,0,0,0,0};
        v8f cz  = cr, cnx = cr, cnh = cr;

#pragma unroll
        for (int k0 = 0; k0 < 64; k0 += 32) {         // K over padded D
            v16bf a  = load_tile_bf16(xt, 64, m0, k0, lane);
            v16bf wr = load_tile_bf16(Wih, 64, 0 * H_ + n0, k0, lane);
            v16bf wz = load_tile_bf16(Wih, 64, 1 * H_ + n0, k0, lane);
            v16bf wn = load_tile_bf16(Wih, 64, 2 * H_ + n0, k0, lane);
            cr  = wmma_bf16(a, wr, cr);
            cz  = wmma_bf16(a, wz, cz);
            cnx = wmma_bf16(a, wn, cnx);
        }
        for (int k0 = 0; k0 < H_; k0 += 32) {         // K over H
            v16bf a  = load_tile_bf16(hprev, H_, m0, k0, lane);
            v16bf wr = load_tile_bf16(Whh, H_, 0 * H_ + n0, k0, lane);
            v16bf wz = load_tile_bf16(Whh, H_, 1 * H_ + n0, k0, lane);
            v16bf wn = load_tile_bf16(Whh, H_, 2 * H_ + n0, k0, lane);
            cr  = wmma_bf16(a, wr, cr);
            cz  = wmma_bf16(a, wz, cz);
            cnh = wmma_bf16(a, wn, cnh);
        }
#pragma unroll
        for (int v = 0; v < 8; ++v) {
            float r = sigmoidf(cr[v] + br_);
            float z = sigmoidf(cz[v] + bz_);
            float n = tanhf(cnx[v] + bnx + r * (cnh[v] + bnh));
            int b = mbase + v;
            float hp = bf2f(hprev[b * H_ + col]);
            float hn = (1.0f - z) * n + z * hp;
            o1t[b * H_ + col]   = hn;
            hnext[b * H_ + col] = f2bf(hn);
        }
        step_barrier(bar, t);
    }
}

// ---------- persistent LSTM ----------
__global__ __launch_bounds__(256) void lstm_persist_kernel(
    const unsigned short* __restrict__ xb,     // [S,64,64] bf16
    unsigned short* __restrict__ h0,
    unsigned short* __restrict__ h1,
    float* __restrict__ cstate,                // [64,512] f32 (owner-tile in place)
    float* __restrict__ o2,                    // [S,64,512] f32
    const unsigned short* __restrict__ Wih,    // [2048,64]
    const unsigned short* __restrict__ Whh,    // [2048,512]
    const float* __restrict__ bih,             // [2048]
    const float* __restrict__ bhh,             // [2048]
    unsigned* __restrict__ bar)
{
    const int lane = threadIdx.x & 31;
    const int wv   = threadIdx.x >> 5;
    const int idx  = blockIdx.x * 8 + wv;
    const int m0   = (idx >> 5) << 4;
    const int n0   = (idx & 31) << 4;

    const int col   = n0 + (lane & 15);
    const float bi_ = bih[col]          + bhh[col];
    const float bf_ = bih[H_ + col]     + bhh[H_ + col];
    const float bg_ = bih[2 * H_ + col] + bhh[2 * H_ + col];
    const float bo_ = bih[3 * H_ + col] + bhh[3 * H_ + col];
    const int mbase = m0 + ((lane >> 4) << 3);

    for (int t = 0; t < S_; ++t) {
        const unsigned short* hprev = (t & 1) ? h1 : h0;
        unsigned short*       hnext = (t & 1) ? h0 : h1;
        const unsigned short* xt = xb + (size_t)t * (B_ * 64);
        float* o2t = o2 + (size_t)t * (B_ * H_);

        if (t + 1 < S_)
            __builtin_prefetch(xb + (size_t)(t + 1) * (B_ * 64) + lane * 128, 0, 0);

        v8f ci = {0,0,0,0,0,0,0,0};
        v8f cf = ci, cg = ci, co = ci;

#pragma unroll
        for (int k0 = 0; k0 < 64; k0 += 32) {
            v16bf a  = load_tile_bf16(xt, 64, m0, k0, lane);
            v16bf wi = load_tile_bf16(Wih, 64, 0 * H_ + n0, k0, lane);
            v16bf wf = load_tile_bf16(Wih, 64, 1 * H_ + n0, k0, lane);
            v16bf wg = load_tile_bf16(Wih, 64, 2 * H_ + n0, k0, lane);
            v16bf wo = load_tile_bf16(Wih, 64, 3 * H_ + n0, k0, lane);
            ci = wmma_bf16(a, wi, ci);
            cf = wmma_bf16(a, wf, cf);
            cg = wmma_bf16(a, wg, cg);
            co = wmma_bf16(a, wo, co);
        }
        for (int k0 = 0; k0 < H_; k0 += 32) {
            v16bf a  = load_tile_bf16(hprev, H_, m0, k0, lane);
            v16bf wi = load_tile_bf16(Whh, H_, 0 * H_ + n0, k0, lane);
            v16bf wf = load_tile_bf16(Whh, H_, 1 * H_ + n0, k0, lane);
            v16bf wg = load_tile_bf16(Whh, H_, 2 * H_ + n0, k0, lane);
            v16bf wo = load_tile_bf16(Whh, H_, 3 * H_ + n0, k0, lane);
            ci = wmma_bf16(a, wi, ci);
            cf = wmma_bf16(a, wf, cf);
            cg = wmma_bf16(a, wg, cg);
            co = wmma_bf16(a, wo, co);
        }
#pragma unroll
        for (int v = 0; v < 8; ++v) {
            float i = sigmoidf(ci[v] + bi_);
            float f = sigmoidf(cf[v] + bf_);
            float g = tanhf(cg[v] + bg_);
            float o = sigmoidf(co[v] + bo_);
            int b = mbase + v;
            float cp = cstate[b * H_ + col];
            float cn = f * cp + i * g;
            cstate[b * H_ + col] = cn;
            float hn = o * tanhf(cn);
            o2t[b * H_ + col]   = hn;
            hnext[b * H_ + col] = f2bf(hn);
        }
        step_barrier(bar, t);
    }
}

// ---------- BatchNorm stats: one block per channel ----------
__global__ __launch_bounds__(256) void bn_stats_kernel(
    const float* __restrict__ o1, float* __restrict__ mean, float* __restrict__ rstd)
{
    const int h = blockIdx.x;
    float s = 0.f, s2 = 0.f;
    for (int r = threadIdx.x; r < BS_; r += 256) {
        float v = o1[(size_t)r * H_ + h];
        s += v; s2 += v * v;
    }
    __shared__ float sh1[256], sh2[256];
    sh1[threadIdx.x] = s; sh2[threadIdx.x] = s2;
    __syncthreads();
    for (int off = 128; off > 0; off >>= 1) {
        if (threadIdx.x < off) {
            sh1[threadIdx.x] += sh1[threadIdx.x + off];
            sh2[threadIdx.x] += sh2[threadIdx.x + off];
        }
        __syncthreads();
    }
    if (threadIdx.x == 0) {
        float m = sh1[0] * (1.0f / BS_);
        float var = sh2[0] * (1.0f / BS_) - m * m;
        mean[h] = m;
        rstd[h] = rsqrtf(var + 1e-5f);
    }
}

// ---------- fused BN + fc1 + fc2 + ssr: one block per (s,b) row ----------
__global__ __launch_bounds__(256) void fc12_kernel(
    const float* __restrict__ o1, const float* __restrict__ mean,
    const float* __restrict__ rstd, const float* __restrict__ gamma,
    const float* __restrict__ beta, const float* __restrict__ w1,
    const float* __restrict__ b1, const float* __restrict__ w2,
    const float* __restrict__ b2, float* __restrict__ out_ssr,
    unsigned short* __restrict__ xbL)
{
    const int row = blockIdx.x;                   // s*B + b
    const float* o = o1 + (size_t)row * H_;
    float a1 = 0.f, a2 = 0.f;
    for (int j = threadIdx.x; j < H_; j += 256) {
        float v = (o[j] - mean[j]) * rstd[j] * gamma[j] + beta[j];
        a1 += v * w1[j];
        a2 += v * w2[j];
    }
    __shared__ float sh1[256], sh2[256];
    sh1[threadIdx.x] = a1; sh2[threadIdx.x] = a2;
    __syncthreads();
    for (int off = 128; off > 0; off >>= 1) {
        if (threadIdx.x < off) {
            sh1[threadIdx.x] += sh1[threadIdx.x + off];
            sh2[threadIdx.x] += sh2[threadIdx.x + off];
        }
        __syncthreads();
    }
    if (threadIdx.x == 0) {
        float sr = fmaxf(sh1[0] + b1[0], 0.0f);
        float w  = fabsf(sigmoidf(sh2[0] + b2[0]));
        float ssr = sr * (w + 1.0f);
        int s = row >> 6, b = row & (B_ - 1);
        out_ssr[(size_t)b * S_ + s] = ssr;        // ssr output [B,S]
        xbL[(size_t)row * 64 + 40] = f2bf(ssr);   // feed LSTM input channel
    }
}

// ---------- fc3 head: one block per (s,b) row ----------
__global__ __launch_bounds__(256) void fc3_kernel(
    const float* __restrict__ o2, const float* __restrict__ w3,
    const float* __restrict__ b3, float* __restrict__ out)
{
    const int row = blockIdx.x;                   // s*B + b
    const float* o = o2 + (size_t)row * H_;
    float a = 0.f;
    for (int j = threadIdx.x; j < H_; j += 256) a += o[j] * w3[j];
    __shared__ float sh[256];
    sh[threadIdx.x] = a;
    __syncthreads();
    for (int off = 128; off > 0; off >>= 1) {
        if (threadIdx.x < off) sh[threadIdx.x] += sh[threadIdx.x + off];
        __syncthreads();
    }
    if (threadIdx.x == 0) {
        int s = row >> 6, b = row & (B_ - 1);
        out[(size_t)b * S_ + s] = fmaxf(sh[0] + b3[0], 0.0f);
    }
}

// ---------- host orchestration ----------
extern "C" void kernel_launch(void* const* d_in, const int* in_sizes, int n_in,
                              void* d_out, int out_size, void* d_ws, size_t ws_size,
                              hipStream_t stream) {
    (void)in_sizes; (void)n_in; (void)out_size; (void)ws_size;
    const float* x        = (const float*)d_in[0];
    const float* gru_Wih  = (const float*)d_in[1];
    const float* gru_Whh  = (const float*)d_in[2];
    const float* gru_bih  = (const float*)d_in[3];
    const float* gru_bhh  = (const float*)d_in[4];
    const float* lstm_Wih = (const float*)d_in[5];
    const float* lstm_Whh = (const float*)d_in[6];
    const float* lstm_bih = (const float*)d_in[7];
    const float* lstm_bhh = (const float*)d_in[8];
    const float* fc1_w = (const float*)d_in[9];
    const float* fc1_b = (const float*)d_in[10];
    const float* fc2_w = (const float*)d_in[11];
    const float* fc2_b = (const float*)d_in[12];
    const float* fc3_w = (const float*)d_in[13];
    const float* fc3_b = (const float*)d_in[14];
    const float* bn_g  = (const float*)d_in[15];
    const float* bn_b  = (const float*)d_in[16];
    float* out = (float*)d_out;

    // workspace carve-up (deterministic, same every call)
    char* ws = (char*)d_ws;
    size_t off = 0;
    auto take = [&](size_t bytes) {
        size_t o = off;
        off += (bytes + 255) & ~(size_t)255;
        return o;
    };
    unsigned short* xbG  = (unsigned short*)(ws + take((size_t)BS_ * 64 * 2));
    unsigned short* xbL  = (unsigned short*)(ws + take((size_t)BS_ * 64 * 2));
    unsigned short* WihG = (unsigned short*)(ws + take((size_t)3 * H_ * 64 * 2));
    unsigned short* WhhG = (unsigned short*)(ws + take((size_t)3 * H_ * H_ * 2));
    unsigned short* WihL = (unsigned short*)(ws + take((size_t)4 * H_ * 64 * 2));
    unsigned short* WhhL = (unsigned short*)(ws + take((size_t)4 * H_ * H_ * 2));
    unsigned short* hG   = (unsigned short*)(ws + take((size_t)2 * B_ * H_ * 2));
    unsigned short* hL   = (unsigned short*)(ws + take((size_t)2 * B_ * H_ * 2));
    float*          cL   = (float*)(ws + take((size_t)B_ * H_ * 4));
    float*          mean = (float*)(ws + take((size_t)H_ * 4));
    float*          rstd = (float*)(ws + take((size_t)H_ * 4));
    unsigned*       bars = (unsigned*)(ws + take(256));          // 2 step counters
    float*          o1   = (float*)(ws + take((size_t)BS_ * H_ * 4)); // aliased as o2

    // 1) weight conversion to bf16 (Wih padded to K=64)
    {
        int n;
        n = 3 * H_ * 64;  cvt_pad_kernel<<<(n + 255) / 256, 256, 0, stream>>>(gru_Wih,  WihG, 3 * H_, D_,     64);
        n = 3 * H_ * H_;  cvt_pad_kernel<<<(n + 255) / 256, 256, 0, stream>>>(gru_Whh,  WhhG, 3 * H_, H_,     H_);
        n = 4 * H_ * 64;  cvt_pad_kernel<<<(n + 255) / 256, 256, 0, stream>>>(lstm_Wih, WihL, 4 * H_, D_ + 1, 64);
        n = 4 * H_ * H_;  cvt_pad_kernel<<<(n + 255) / 256, 256, 0, stream>>>(lstm_Whh, WhhL, 4 * H_, H_,     H_);
    }
    // 2) padded time-major bf16 inputs
    {
        int n = BS_ * 64;
        build_x_kernel<<<(n + 255) / 256, 256, 0, stream>>>(x, xbG, xbL);
    }
    // 3) zero recurrent state (both h parity buffers, c, barrier counters)
    {
        int n;
        n = 2 * B_ * H_ / 2; zero32_kernel<<<(n + 255) / 256, 256, 0, stream>>>((unsigned int*)hG, n);
        n = 2 * B_ * H_ / 2; zero32_kernel<<<(n + 255) / 256, 256, 0, stream>>>((unsigned int*)hL, n);
        n = B_ * H_;         zero32_kernel<<<(n + 255) / 256, 256, 0, stream>>>((unsigned int*)cL, n);
        zero32_kernel<<<1, 64, 0, stream>>>((unsigned int*)bars, 64);
    }
    // 4) GRU recurrence: one persistent kernel, in-kernel step barrier
    gru_persist_kernel<<<NBLK_, 256, 0, stream>>>(
        xbG, hG, hG + (size_t)B_ * H_, o1, WihG, WhhG, gru_bih, gru_bhh, bars + 0);

    // 5) BatchNorm stats + fused BN/fc1/fc2 -> ssr (and LSTM input channel)
    bn_stats_kernel<<<H_, 256, 0, stream>>>(o1, mean, rstd);
    fc12_kernel<<<BS_, 256, 0, stream>>>(o1, mean, rstd, bn_g, bn_b,
                                         fc1_w, fc1_b, fc2_w, fc2_b,
                                         out + BS_, xbL);

    // 6) LSTM recurrence: persistent kernel; o2 aliases o1 (o1 dead after fc12)
    float* o2 = o1;
    lstm_persist_kernel<<<NBLK_, 256, 0, stream>>>(
        xbL, hL, hL + (size_t)B_ * H_, cL, o2, WihL, WhhL, lstm_bih, lstm_bhh, bars + 64);

    // 7) fc3 head -> out [B,S]
    fc3_kernel<<<BS_, 256, 0, stream>>>(o2, fc3_w, fc3_b, out);
}
